// RecallLSTM_66924180406293
// MI455X (gfx1250) — matrix operand order repaired and checked by reading, not compiled, not run on hardware
//
#include <hip/hip_runtime.h>

typedef __attribute__((ext_vector_type(16))) _Float16 v16h;
typedef __attribute__((ext_vector_type(8)))  _Float16 v8h;
typedef __attribute__((ext_vector_type(8)))  float    v8f;

namespace {
constexpr int kH = 64;          // hidden
constexpr int kB = 256;         // batch
constexpr int kS = 2048;        // seq len
constexpr int kI = 32;          // layer-0 input size
constexpr int kG = 4 * kH;      // 256 gate columns (i,f,g,o)
constexpr int kBM = 16;         // batch rows per workgroup (WMMA M)
constexpr int kThreads = 128;   // 4 waves; wave w owns gate cols 16w..16w+15 of each gate
}

// gfx1250 hardware tanh (TRANS32 pipe, co-executes with XDL/WMMA).
__device__ __forceinline__ float tanh_hw(float x) {
#if __has_builtin(__builtin_amdgcn_tanhf)
  return __builtin_amdgcn_tanhf(x);
#else
  float r;
  asm("v_tanh_f32 %0, %1" : "=v"(r) : "v"(x));
  return r;
#endif
}
__device__ __forceinline__ float sigmoid_f(float x) {
  // sigmoid(x) = 0.5 + 0.5*tanh(x/2): 1 TRANS + 2 VALU
  return __builtin_fmaf(0.5f, tanh_hw(0.5f * x), 0.5f);
}

// A-operand (16x32 f16): lane lo = row M; elems e<8 -> K = 8*hi + e,
// e>=8 -> K = 16 + 8*hi + (e-8).  Two contiguous 8-half LDS chunks.
__device__ __forceinline__ v16h make_a(const _Float16* row, int hi) {
  v8h p0 = *(const v8h*)(row + hi * 8);
  v8h p1 = *(const v8h*)(row + 16 + hi * 8);
  v16h a;
#pragma unroll
  for (int e = 0; e < 8; ++e) { a[e] = p0[e]; a[8 + e] = p1[e]; }
  return a;
}

// B-operand (32x16 f16): lane lo = col N; elems e<8 -> K = 8*hi + e,
// e>=8 -> K = 16 + 8*hi + (e-8).  W stored row-major [n][k] -> two 8-half chunks.
__device__ __forceinline__ v16h make_b(const _Float16* wrow, int hi) {
  v8h p0 = *(const v8h*)(wrow + hi * 8);
  v8h p1 = *(const v8h*)(wrow + 16 + hi * 8);
  v16h b;
#pragma unroll
  for (int e = 0; e < 8; ++e) { b[e] = p0[e]; b[8 + e] = p1[e]; }
  return b;
}

__device__ __forceinline__ v8f wmma_f16(v16h a, v16h b, v8f c) {
  return __builtin_amdgcn_wmma_f32_16x16x32_f16(
      /*neg_a=*/false, a, /*neg_b=*/false, b,
      /*c_mod=*/(short)0, c, /*reuse_a=*/false, /*reuse_b=*/false);
}

__global__ __launch_bounds__(kThreads, 1)
void lstm2_fused_kernel(const float* __restrict__ x,
                        const float* __restrict__ h0in,
                        const float* __restrict__ c0in,
                        const float* __restrict__ Wih0,
                        const float* __restrict__ Whh0,
                        const float* __restrict__ bih0,
                        const float* __restrict__ bhh0,
                        const float* __restrict__ Wih1,
                        const float* __restrict__ Whh1,
                        const float* __restrict__ bih1,
                        const float* __restrict__ bhh1,
                        const float* __restrict__ Wh,
                        const float* __restrict__ bh,
                        float* __restrict__ out) {
  // ---- LDS: f16 weight staging (only used to build register operands)
  //      + ping-pong h / x tiles ----
  __shared__ alignas(32) _Float16 sWih0[kG][kI];       // 16 KB
  __shared__ alignas(32) _Float16 sWhh0[kG][kH];       // 32 KB
  __shared__ alignas(32) _Float16 sWih1[kG][kH];       // 32 KB
  __shared__ alignas(32) _Float16 sWhh1[kG][kH];       // 32 KB
  __shared__ alignas(32) _Float16 sXt[2][kBM][kI];     // ping-pong x tile (f16)
  __shared__ alignas(32) _Float16 sH0h[2][kBM][kH];    // ping-pong h0 (f16)
  __shared__ alignas(32) _Float16 sH1h[2][kBM][kH];    // ping-pong h1 (f16)

  const int tid  = threadIdx.x;
  const int wave = tid >> 5;       // 0..3
  const int lane = tid & 31;
  const int lo   = lane & 15;
  const int hi   = lane >> 4;
  const int bbase = blockIdx.x * kBM;
  const int jcol = (wave << 4) + lo;   // hidden column owned by this lane: 0..63
  const int mrow0 = 8 * hi;            // accumulator rows: mrow0..mrow0+7

  // ---- one-time weight staging (f32 -> f16 in LDS) ----
  for (int i = tid; i < kG * kI; i += kThreads)
    sWih0[i / kI][i % kI] = (_Float16)Wih0[i];
  for (int i = tid; i < kG * kH; i += kThreads) {
    int r = i / kH, cc = i % kH;
    sWhh0[r][cc] = (_Float16)Whh0[i];
    sWih1[r][cc] = (_Float16)Wih1[i];
    sWhh1[r][cc] = (_Float16)Whh1[i];
  }
  const float bhv = bh[0];

  // ---- per-lane register state: biases, c, h(f32), head weights ----
  float rb0[4], rb1[4];
#pragma unroll
  for (int g = 0; g < 4; ++g) {
    int n = (g << 6) + jcol;
    rb0[g] = bih0[n] + bhh0[n];
    rb1[g] = bih1[n] + bhh1[n];
  }
  float whd[8];
  {
    const int jc = (tid & 7) * 8;
#pragma unroll
    for (int e = 0; e < 8; ++e) whd[e] = Wh[jc + e];
  }
  float c0r[8], c1r[8], h0r[8], h1r[8];
#pragma unroll
  for (int r = 0; r < 8; ++r) {
    int m = mrow0 + r, b = bbase + m;
    h0r[r] = h0in[0 * kB * kH + b * kH + jcol];
    h1r[r] = h0in[1 * kB * kH + b * kH + jcol];
    c0r[r] = c0in[0 * kB * kH + b * kH + jcol];
    c1r[r] = c0in[1 * kB * kH + b * kH + jcol];
    sH0h[0][m][jcol] = (_Float16)h0r[r];
    sH1h[0][m][jcol] = (_Float16)h1r[r];
  }

  // ---- x prefetch: each thread owns 4 elements of the 16x32 tile ----
  const int xi = tid & 31;
  const int xm = tid >> 5;                 // rows xm, xm+4, xm+8, xm+12
  const float* xrow[4];
  float xr[4];
#pragma unroll
  for (int k = 0; k < 4; ++k) {
    xrow[k] = x + (size_t)(bbase + xm + 4 * k) * kS * kI + xi;
    xr[k] = xrow[k][0];
  }
  __syncthreads();   // weights + initial h staged

  // ---- hoist ALL loop-invariant B-operands into registers (224 VGPRs) ----
  v16h bWih0[4];        // x projection, K=32 (one chunk)
  v16h bWhh0[4][2];     // recurrent l0, K=64 (two chunks)
  v16h bWih1[4][2];     // l0->l1, K=64
  v16h bWhh1[4][2];     // recurrent l1, K=64
#pragma unroll
  for (int g = 0; g < 4; ++g) {
    const int n = (g << 6) + jcol;
    bWih0[g]    = make_b(&sWih0[n][0],  hi);
    bWhh0[g][0] = make_b(&sWhh0[n][0],  hi);
    bWhh0[g][1] = make_b(&sWhh0[n][32], hi);
    bWih1[g][0] = make_b(&sWih1[n][0],  hi);
    bWih1[g][1] = make_b(&sWih1[n][32], hi);
    bWhh1[g][0] = make_b(&sWhh1[n][0],  hi);
    bWhh1[g][1] = make_b(&sWhh1[n][32], hi);
  }

  // ---- prologue of the x-projection software pipeline ----
  // stage x(0) into buffer 0, compute xacc = x(0)@Wih0^T, prefetch x(1)
#pragma unroll
  for (int k = 0; k < 4; ++k) sXt[0][xm + 4 * k][xi] = (_Float16)xr[k];
  __syncthreads();
  v8f xacc[4];
  {
    v16h ax = make_a(&sXt[0][lo][0], hi);
#pragma unroll
    for (int g = 0; g < 4; ++g) {
      v8f z = {};
      xacc[g] = wmma_f16(ax, bWih0[g], z);
    }
  }
#pragma unroll
  for (int k = 0; k < 4; ++k) xr[k] = xrow[k][(size_t)((1 < kS) ? 1 : 0) * kI];

  for (int t = 0; t < kS; ++t) {
    const int p = t & 1;        // read recurrent h from buffer p, write to q
    const int q = 1 - p;

    // ---- phase A: recurrent GEMMs.  Layer-0 gates start from the
    //      precomputed x projection; layer-1's Whh1 half only needs
    //      h1(t-1) so it issues now and overlaps layer-0 elementwise ----
    v16h aha = make_a(&sH0h[p][lo][0], hi);
    v16h ahb = make_a(&sH0h[p][lo][32], hi);
    v16h a1a = make_a(&sH1h[p][lo][0], hi);
    v16h a1b = make_a(&sH1h[p][lo][32], hi);
    v8f accA[4], accB[4];
#pragma unroll
    for (int g = 0; g < 4; ++g) {
      v8f a = xacc[g];
      a = wmma_f16(aha, bWhh0[g][0], a);
      a = wmma_f16(ahb, bWhh0[g][1], a);
      accA[g] = a;
      v8f bacc = {};
      bacc = wmma_f16(a1a, bWhh1[g][0], bacc);
      bacc = wmma_f16(a1b, bWhh1[g][1], bacc);
      accB[g] = bacc;
    }

    // ---- phase B: layer-0 elementwise (registers only, hw tanh) ----
#pragma unroll
    for (int r = 0; r < 8; ++r) {
      float gi = accA[0][r] + rb0[0];
      float gf = accA[1][r] + rb0[1];
      float gg = accA[2][r] + rb0[2];
      float go = accA[3][r] + rb0[3];
      float cv = sigmoid_f(gf) * c0r[r] + sigmoid_f(gi) * tanh_hw(gg);
      float hv = sigmoid_f(go) * tanh_hw(cv);
      c0r[r] = cv;
      h0r[r] = hv;
      sH0h[q][mrow0 + r][jcol] = (_Float16)hv;
    }
    __syncthreads();

    // ---- phase D: finish layer-1 gates with h0(t)@Wih1^T ----
    {
      v16h a0a = make_a(&sH0h[q][lo][0], hi);
      v16h a0b = make_a(&sH0h[q][lo][32], hi);
#pragma unroll
      for (int g = 0; g < 4; ++g) {
        v8f bacc = accB[g];
        bacc = wmma_f16(a0a, bWih1[g][0], bacc);
        bacc = wmma_f16(a0b, bWih1[g][1], bacc);
        accB[g] = bacc;
      }
    }
    // ---- phase E: layer-1 elementwise ----
#pragma unroll
    for (int r = 0; r < 8; ++r) {
      float gi = accB[0][r] + rb1[0];
      float gf = accB[1][r] + rb1[1];
      float gg = accB[2][r] + rb1[2];
      float go = accB[3][r] + rb1[3];
      float cv = sigmoid_f(gf) * c1r[r] + sigmoid_f(gi) * tanh_hw(gg);
      float hv = sigmoid_f(go) * tanh_hw(cv);
      c1r[r] = cv;
      h1r[r] = hv;
      sH1h[q][mrow0 + r][jcol] = (_Float16)hv;
    }

    // ---- phase F: stage x(t+1) (held in xr) and prefetch x(t+2) ----
#pragma unroll
    for (int k = 0; k < 4; ++k) sXt[q][xm + 4 * k][xi] = (_Float16)xr[k];
    {
      int tn = (t + 2 < kS) ? (t + 2) : (kS - 1);
#pragma unroll
      for (int k = 0; k < 4; ++k) xr[k] = xrow[k][(size_t)tn * kI];
    }
    __syncthreads();

    // ---- phase H: head (8 threads/row) + next step's x projection ----
    {
      const int row = tid >> 3;          // 0..15
      const int jc  = (tid & 7) * 8;     // 8-wide contiguous chunk
      v8h hv = *(const v8h*)(&sH1h[q][row][jc]);
      float partial = 0.0f;
#pragma unroll
      for (int e = 0; e < 8; ++e) partial += (float)hv[e] * whd[e];
      partial += __shfl_down(partial, 4, 8);
      partial += __shfl_down(partial, 2, 8);
      partial += __shfl_down(partial, 1, 8);
      if ((tid & 7) == 0)
        out[(size_t)(bbase + row) * kS + t] = sigmoid_f(partial + bhv);
    }
    {
      v16h ax = make_a(&sXt[q][lo][0], hi);
#pragma unroll
      for (int g = 0; g < 4; ++g) {
        v8f z = {};
        xacc[g] = wmma_f16(ax, bWih0[g], z);
      }
    }
  }

  // ---- final states straight from f32 registers ----
  const size_t PO = (size_t)kB * kS;
#pragma unroll
  for (int r = 0; r < 8; ++r) {
    int b = bbase + mrow0 + r;
    out[PO + 0 * kB * kH + b * kH + jcol] = h0r[r];
    out[PO + 1 * kB * kH + b * kH + jcol] = h1r[r];
    out[PO + 2 * kB * kH + b * kH + jcol] = c0r[r];
    out[PO + 3 * kB * kH + b * kH + jcol] = c1r[r];
  }
}

extern "C" void kernel_launch(void* const* d_in, const int* in_sizes, int n_in,
                              void* d_out, int out_size, void* d_ws, size_t ws_size,
                              hipStream_t stream) {
  (void)in_sizes; (void)n_in; (void)out_size; (void)d_ws; (void)ws_size;
  const float* x    = (const float*)d_in[0];
  const float* h0   = (const float*)d_in[1];
  const float* c0   = (const float*)d_in[2];
  const float* Wih0 = (const float*)d_in[3];
  const float* Whh0 = (const float*)d_in[4];
  const float* bih0 = (const float*)d_in[5];
  const float* bhh0 = (const float*)d_in[6];
  const float* Wih1 = (const float*)d_in[7];
  const float* Whh1 = (const float*)d_in[8];
  const float* bih1 = (const float*)d_in[9];
  const float* bhh1 = (const float*)d_in[10];
  const float* Wh   = (const float*)d_in[11];
  const float* bh   = (const float*)d_in[12];
  float* out = (float*)d_out;

  dim3 grid(kB / kBM);      // 16 workgroups, one 16-row batch slice each
  dim3 block(kThreads);     // 4 wave32 waves (1 per SIMD32 -> whole VGPR file)
  hipLaunchKernelGGL(lstm2_fused_kernel, grid, block, 0, stream,
                     x, h0, c0, Wih0, Whh0, bih0, bhh0,
                     Wih1, Whh1, bih1, bhh1, Wh, bh, out);
}
